// CausalWanSelfAttention_85332410237354
// MI455X (gfx1250) — matrix operand order repaired and checked by reading, not compiled, not run on hardware
//
#include <hip/hip_runtime.h>
#include <hip/hip_bf16.h>
#include <math.h>

#define DIM 1536
#define HEADS 12
#define HD 128
#define FRAME 780
#define NFRAMES 13
#define SEQLEN (FRAME * NFRAMES)      /* 10140 */
#define BLOCKQ (FRAME * 2)            /* 1560 */
#define WINDOW (4 * FRAME)            /* 3120 */

typedef __attribute__((ext_vector_type(16))) __bf16 v16bf;
typedef __attribute__((ext_vector_type(8)))  __bf16 v8bf;
typedef __attribute__((ext_vector_type(8)))  float  v8f;
typedef __attribute__((ext_vector_type(4)))  unsigned int u32x4;
typedef __attribute__((ext_vector_type(8)))  int i32x8;
typedef __attribute__((ext_vector_type(4)))  int i32x4;

#if __has_builtin(__builtin_amdgcn_tensor_load_to_lds)
#define HAVE_TDM 1
#else
#define HAVE_TDM 0
#endif

static __device__ __forceinline__ v16bf cat16(v8bf lo, v8bf hi) {
  v16bf r;
#pragma unroll
  for (int j = 0; j < 8; ++j) { r[j] = lo[j]; r[8 + j] = hi[j]; }
  return r;
}

#if HAVE_TDM
// Issue a TDM 2D tile load: tile_x contiguous bf16 elems per row, tile_y rows,
// row stride = stride_elems (bf16 elems), packed row-major into LDS at lds_addr.
// valid_y rows are in-bounds; TDM returns zero for y >= valid_y.
static __device__ __forceinline__ void tdm_load_2d(unsigned int lds_addr, const void* gptr,
                                                   int tile_x, int tile_y, int valid_y,
                                                   int stride_elems) {
  unsigned long long ga = (unsigned long long)(uintptr_t)gptr;
  u32x4 g0;
  g0[0] = 1u;                                            // count=1, user mode
  g0[1] = lds_addr;                                      // lds_addr [63:32]
  g0[2] = (unsigned int)ga;                              // global_addr lo
  g0[3] = (unsigned int)((ga >> 32) & 0x1FFFFFFull) | (2u << 30);  // addr hi + type=2
  i32x8 g1;
  g1[0] = 0x00010000;                                    // data_size=1 (2 bytes)
  g1[1] = (int)(((unsigned)tile_x & 0xFFFFu) << 16);     // tensor_dim0[15:0] @ [63:48]
  g1[2] = (int)((((unsigned)tile_x >> 16) & 0xFFFFu) |   // tensor_dim0[31:16]
                (((unsigned)valid_y & 0xFFFFu) << 16));  // tensor_dim1[15:0]
  g1[3] = (int)((((unsigned)valid_y >> 16) & 0xFFFFu) |  // tensor_dim1[31:16]
                (((unsigned)tile_x & 0xFFFFu) << 16));   // tile_dim0
  g1[4] = (int)((unsigned)tile_y & 0xFFFFu);             // tile_dim1 (tile_dim2=0)
  g1[5] = stride_elems;                                  // tensor_dim0_stride lo32
  g1[6] = 0;                                             // stride0 hi / stride1 lo
  g1[7] = 0;
  i32x4 gz = {0, 0, 0, 0};
#if defined(__clang_major__) && __clang_major__ >= 23
  i32x8 gz8 = {0, 0, 0, 0, 0, 0, 0, 0};
  __builtin_amdgcn_tensor_load_to_lds(g0, g1, gz, gz, gz8, 0);
#else
  __builtin_amdgcn_tensor_load_to_lds(g0, g1, gz, gz, 0);
#endif
}
#endif

// ---------------- conversion kernels ----------------
__global__ void cvt_x_kernel(const float* __restrict__ x, __bf16* __restrict__ xb, int n) {
  int i = blockIdx.x * blockDim.x + threadIdx.x;
  if (i < n) xb[i] = (__bf16)x[i];
}

__global__ void cvt_wT_kernel(const float* __restrict__ Wq, const float* __restrict__ Wk,
                              const float* __restrict__ Wv, const float* __restrict__ Wo,
                              __bf16* __restrict__ tq, __bf16* __restrict__ tk,
                              __bf16* __restrict__ tv, __bf16* __restrict__ to) {
  const int n = blockIdx.x;
  const float* W; __bf16* T;
  switch (blockIdx.y) {
    case 0:  W = Wq; T = tq; break;
    case 1:  W = Wk; T = tk; break;
    case 2:  W = Wv; T = tv; break;
    default: W = Wo; T = to; break;
  }
  for (int kk = threadIdx.x; kk < DIM; kk += blockDim.x)
    T[(size_t)n * DIM + kk] = (__bf16)W[(size_t)kk * DIM + n];
}

// ---------------- bf16 WMMA GEMM: out = A(LxDIM) * Wt^T + bias ----------------
template <bool BF16OUT>
__global__ void __launch_bounds__(256)
gemm_bias_kernel(const __bf16* __restrict__ A, const __bf16* __restrict__ Wt,
                 const float* __restrict__ bias, void* __restrict__ outp) {
  const int mbase = blockIdx.x * 64;
  const int nbase = blockIdx.y * 64;
  const int t = threadIdx.x;
  const int lane = t & 31;
  const int wave = t >> 5;
  const int wm = (wave & 3) * 16;
  const int wn = (wave >> 2) * 32;
  const int lrow = lane & 15, lhi = lane >> 4;
  v8f acc0 = {}, acc1 = {};

#if HAVE_TDM
  // double-buffered TDM staging
  __shared__ alignas(16) __bf16 As[2][64][32];
  __shared__ alignas(16) __bf16 Bs[2][64][32];
  const __bf16* Abase = A + (size_t)mbase * DIM;
  const __bf16* Bbase = Wt + (size_t)nbase * DIM;
  const int mvalid = min(64, SEQLEN - mbase);
  if (wave == 0) {
    tdm_load_2d((unsigned int)(uintptr_t)&As[0][0][0], Abase, 32, 64, mvalid, DIM);
    tdm_load_2d((unsigned int)(uintptr_t)&Bs[0][0][0], Bbase, 32, 64, 64, DIM);
  }
  int buf = 0;
  for (int k0 = 0; k0 < DIM; k0 += 32, buf ^= 1) {
    if (wave == 0) __builtin_amdgcn_s_wait_tensorcnt(0);
    __syncthreads();
    if (wave == 0 && k0 + 32 < DIM) {
      tdm_load_2d((unsigned int)(uintptr_t)&As[buf ^ 1][0][0], Abase + (k0 + 32), 32, 64, mvalid, DIM);
      tdm_load_2d((unsigned int)(uintptr_t)&Bs[buf ^ 1][0][0], Bbase + (k0 + 32), 32, 64, 64, DIM);
    }
    const int ab = lhi * 8;
    v16bf af = cat16(*(const v8bf*)&As[buf][wm + lrow][ab],
                     *(const v8bf*)&As[buf][wm + lrow][ab + 16]);
    const int kb = lhi * 8;
    v16bf b0 = cat16(*(const v8bf*)&Bs[buf][wn + lrow][kb],
                     *(const v8bf*)&Bs[buf][wn + lrow][kb + 16]);
    v16bf b1 = cat16(*(const v8bf*)&Bs[buf][wn + 16 + lrow][kb],
                     *(const v8bf*)&Bs[buf][wn + 16 + lrow][kb + 16]);
    acc0 = __builtin_amdgcn_wmma_f32_16x16x32_bf16(false, af, false, b0, (short)0, acc0, false, false);
    acc1 = __builtin_amdgcn_wmma_f32_16x16x32_bf16(false, af, false, b1, (short)0, acc1, false, false);
    __syncthreads();
  }
#else
  // manual LDS staging fallback
  __shared__ alignas(16) __bf16 As[64][32];
  __shared__ alignas(16) __bf16 Bs[64][32];
  const int lr = t >> 2;
  const int lc = (t & 3) * 8;
  const int arow = min(mbase + lr, SEQLEN - 1);
  const int brow = nbase + lr;
  for (int k0 = 0; k0 < DIM; k0 += 32) {
    *(v8bf*)&As[lr][lc] = *(const v8bf*)&A[(size_t)arow * DIM + k0 + lc];
    *(v8bf*)&Bs[lr][lc] = *(const v8bf*)&Wt[(size_t)brow * DIM + k0 + lc];
    if (k0 + 32 < DIM) {
      __builtin_prefetch(&A[(size_t)arow * DIM + k0 + 32 + lc], 0, 1);
      __builtin_prefetch(&Wt[(size_t)brow * DIM + k0 + 32 + lc], 0, 1);
    }
    __syncthreads();
    const int ab = lhi * 8;
    v16bf af = cat16(*(const v8bf*)&As[wm + lrow][ab],
                     *(const v8bf*)&As[wm + lrow][ab + 16]);
    const int kb = lhi * 8;
    v16bf b0 = cat16(*(const v8bf*)&Bs[wn + lrow][kb],
                     *(const v8bf*)&Bs[wn + lrow][kb + 16]);
    v16bf b1 = cat16(*(const v8bf*)&Bs[wn + 16 + lrow][kb],
                     *(const v8bf*)&Bs[wn + 16 + lrow][kb + 16]);
    acc0 = __builtin_amdgcn_wmma_f32_16x16x32_bf16(false, af, false, b0, (short)0, acc0, false, false);
    acc1 = __builtin_amdgcn_wmma_f32_16x16x32_bf16(false, af, false, b1, (short)0, acc1, false, false);
    __syncthreads();
  }
#endif

  const int col0 = nbase + wn + lrow;
  const float bv0 = bias[col0];
  const float bv1 = bias[col0 + 16];
#pragma unroll
  for (int i = 0; i < 8; ++i) {
    int m = mbase + wm + i + 8 * lhi;
    if (m < SEQLEN) {
      float r0 = acc0[i] + bv0;
      float r1 = acc1[i] + bv1;
      if constexpr (BF16OUT) {
        ((__bf16*)outp)[(size_t)m * DIM + col0]      = (__bf16)r0;
        ((__bf16*)outp)[(size_t)m * DIM + col0 + 16] = (__bf16)r1;
      } else {
        ((float*)outp)[(size_t)m * DIM + col0]      = r0;
        ((float*)outp)[(size_t)m * DIM + col0 + 16] = r1;
      }
    }
  }
}

// ---------------- fused RMSNorm + RoPE (in place on bf16 q/k) ----------------
__global__ void __launch_bounds__(256)
norm_rope_kernel(__bf16* __restrict__ q, __bf16* __restrict__ k,
                 const float* __restrict__ gq, const float* __restrict__ gk,
                 const float* __restrict__ freqs) {
  __shared__ float row[DIM];
  __shared__ float red[8];
  const int s = blockIdx.x;
  __bf16* buf = blockIdx.y ? k : q;
  const float* g = blockIdx.y ? gk : gq;
  const int t = threadIdx.x;

  float ss = 0.f;
#pragma unroll
  for (int j = 0; j < DIM / 256; ++j) {
    float v = (float)buf[(size_t)s * DIM + t + j * 256];
    row[t + j * 256] = v;
    ss += v * v;
  }
  for (int o = 1; o < 32; o <<= 1) ss += __shfl_xor(ss, o, 32);
  if ((t & 31) == 0) red[t >> 5] = ss;
  __syncthreads();
  float tot = 0.f;
#pragma unroll
  for (int w = 0; w < 8; ++w) tot += red[w];
  const float r = rsqrtf(tot / (float)DIM + 1e-6f);

#pragma unroll
  for (int j = 0; j < 3; ++j) {
    int pair = t + j * 256;
    int h = pair >> 6;
    int p = pair & 63;
    int e = h * HD + 2 * p;
    float xr = row[e]     * r * g[e];
    float xi = row[e + 1] * r * g[e + 1];
    float ang = freqs[(size_t)s * (HD / 2) + p];
    float c = __cosf(ang), sn = __sinf(ang);
    buf[(size_t)s * DIM + e]     = (__bf16)(xr * c - xi * sn);
    buf[(size_t)s * DIM + e + 1] = (__bf16)(xr * sn + xi * c);
  }
}

// ---------------- flash attention: one wave per (16-query tile, head) ----------------
static __device__ __forceinline__ void kv_range(int qi, int& lo, int& hi) {
  if (qi < FRAME) { lo = 0; hi = FRAME; }
  else {
    int blk = (qi - FRAME) / BLOCKQ;
    int qe = min(FRAME + (blk + 1) * BLOCKQ, SEQLEN);
    hi = qe;
    lo = max(0, qe - WINDOW);
  }
}

__global__ void __launch_bounds__(32)
attn_kernel(const __bf16* __restrict__ q, const __bf16* __restrict__ k,
            const __bf16* __restrict__ v, __bf16* __restrict__ o) {
  __shared__ alignas(16) __bf16 Vt[HD][32];
  __shared__ alignas(16) __bf16 Ps[16][32];
  const int qbase = blockIdx.x * 16;
  const int h = blockIdx.y;
  const int lane = threadIdx.x;
  const int lrow = lane & 15, lhi = lane >> 4;
  const float rscale = 0.08838834764831845f;  // 1/sqrt(128)
  const float NEG = -__builtin_inff();

  int lo0, hi0, lo1, hi1;
  kv_range(min(qbase, SEQLEN - 1), lo0, hi0);
  kv_range(min(qbase + 15, SEQLEN - 1), lo1, hi1);
  const int kvlo = min(lo0, lo1);
  const int kvhi = max(hi0, hi1);

  const int qrow = min(qbase + lrow, SEQLEN - 1);
  const size_t qoff = (size_t)qrow * DIM + h * HD;
  v16bf qf[4];
#pragma unroll
  for (int ks = 0; ks < 4; ++ks) {
    int d0 = ks * 32 + lhi * 8;
    qf[ks] = cat16(*(const v8bf*)&q[qoff + d0], *(const v8bf*)&q[qoff + d0 + 16]);
  }
  int rlo[8], rhi[8];
#pragma unroll
  for (int i = 0; i < 8; ++i) {
    int qi = min(qbase + i + 8 * lhi, SEQLEN - 1);
    kv_range(qi, rlo[i], rhi[i]);
  }
  float mrow[8], lsum[8];
  v8f Oc[8];
  v8f zero = {};
#pragma unroll
  for (int i = 0; i < 8; ++i) { mrow[i] = NEG; lsum[i] = 0.f; }
#pragma unroll
  for (int dt = 0; dt < 8; ++dt) Oc[dt] = zero;

  for (int kv = kvlo; kv < kvhi; kv += 32) {
    // ---- batch-load all 8 K fragments, then 8 WMMAs ----
    v8f s0 = zero, s1 = zero;
    {
      const int kb = lhi * 8;
      const size_t ko0 = (size_t)min(kv + lrow,      SEQLEN - 1) * DIM + h * HD;
      const size_t ko1 = (size_t)min(kv + 16 + lrow, SEQLEN - 1) * DIM + h * HD;
      v16bf kf0[4], kf1[4];
#pragma unroll
      for (int ks = 0; ks < 4; ++ks) {
        int d0 = ks * 32 + kb;
        kf0[ks] = cat16(*(const v8bf*)&k[ko0 + d0], *(const v8bf*)&k[ko0 + d0 + 16]);
        kf1[ks] = cat16(*(const v8bf*)&k[ko1 + d0], *(const v8bf*)&k[ko1 + d0 + 16]);
      }
#pragma unroll
      for (int ks = 0; ks < 4; ++ks) {
        s0 = __builtin_amdgcn_wmma_f32_16x16x32_bf16(false, qf[ks], false, kf0[ks], (short)0, s0, false, false);
        s1 = __builtin_amdgcn_wmma_f32_16x16x32_bf16(false, qf[ks], false, kf1[ks], (short)0, s1, false, false);
      }
    }
    // ---- scale + window mask ----
    const int c0 = kv + lrow;
    const int c1 = kv + 16 + lrow;
#pragma unroll
    for (int i = 0; i < 8; ++i) {
      float e0 = s0[i] * rscale, e1 = s1[i] * rscale;
      if (c0 < rlo[i] || c0 >= rhi[i]) e0 = NEG;
      if (c1 < rlo[i] || c1 >= rhi[i]) e1 = NEG;
      s0[i] = e0; s1[i] = e1;
    }
    // ---- online softmax + write P to LDS ----
#pragma unroll
    for (int i = 0; i < 8; ++i) {
      float mx = fmaxf(s0[i], s1[i]);
      for (int off = 1; off < 16; off <<= 1) mx = fmaxf(mx, __shfl_xor(mx, off, 32));
      float mn = fmaxf(mrow[i], mx);
      bool dead = (mn == NEG);
      float corr = dead ? 1.f : __expf(mrow[i] - mn);
      float p0 = dead ? 0.f : __expf(s0[i] - mn);
      float p1 = dead ? 0.f : __expf(s1[i] - mn);
      float ps = p0 + p1;
      for (int off = 1; off < 16; off <<= 1) ps += __shfl_xor(ps, off, 32);
      lsum[i] = lsum[i] * corr + ps;
      mrow[i] = mn;
#pragma unroll
      for (int dt = 0; dt < 8; ++dt) Oc[dt][i] *= corr;
      Ps[i + 8 * lhi][lrow]      = (__bf16)p0;
      Ps[i + 8 * lhi][16 + lrow] = (__bf16)p1;
    }
    // ---- transpose V chunk into LDS: Vt[d][kv] ----
    {
      const size_t vo = (size_t)min(kv + lane, SEQLEN - 1) * DIM + h * HD;
#pragma unroll
      for (int db = 0; db < HD / 8; ++db) {
        v8bf vv = *(const v8bf*)&v[vo + db * 8];
#pragma unroll
        for (int j = 0; j < 8; ++j) Vt[db * 8 + j][lane] = vv[j];
      }
    }
    __syncthreads();
    // ---- O += P V : batch Vt fragments in groups of 4 ----
    v16bf pf;
    {
      int pb = lhi * 8;
      pf = cat16(*(const v8bf*)&Ps[lrow][pb], *(const v8bf*)&Ps[lrow][pb + 16]);
    }
    const int kp = lhi * 8;
#pragma unroll
    for (int dg = 0; dg < 2; ++dg) {
      v16bf bv[4];
#pragma unroll
      for (int dj = 0; dj < 4; ++dj) {
        int dn = (dg * 4 + dj) * 16 + lrow;
        bv[dj] = cat16(*(const v8bf*)&Vt[dn][kp], *(const v8bf*)&Vt[dn][kp + 16]);
      }
#pragma unroll
      for (int dj = 0; dj < 4; ++dj)
        Oc[dg * 4 + dj] = __builtin_amdgcn_wmma_f32_16x16x32_bf16(false, pf, false, bv[dj], (short)0, Oc[dg * 4 + dj], false, false);
    }
    __syncthreads();
  }
  // ---- normalize + write out ----
#pragma unroll
  for (int i = 0; i < 8; ++i) {
    int qi = qbase + i + 8 * lhi;
    if (qi >= SEQLEN) continue;
    float inv = lsum[i] > 0.f ? 1.f / lsum[i] : 0.f;
    size_t oo = (size_t)qi * DIM + h * HD;
#pragma unroll
    for (int dt = 0; dt < 8; ++dt)
      o[oo + dt * 16 + lrow] = (__bf16)(Oc[dt][i] * inv);
  }
}

// ---------------- launch ----------------
extern "C" void kernel_launch(void* const* d_in, const int* in_sizes, int n_in,
                              void* d_out, int out_size, void* d_ws, size_t ws_size,
                              hipStream_t stream) {
  const float* x     = (const float*)d_in[0];
  const float* freqs = (const float*)d_in[1];
  const float* Wq = (const float*)d_in[2];  const float* bq = (const float*)d_in[3];
  const float* Wk = (const float*)d_in[4];  const float* bk = (const float*)d_in[5];
  const float* Wv = (const float*)d_in[6];  const float* bv = (const float*)d_in[7];
  const float* Wo = (const float*)d_in[8];  const float* bo = (const float*)d_in[9];
  const float* gq = (const float*)d_in[10]; const float* gk = (const float*)d_in[11];
  float* out = (float*)d_out;

  char* w = (char*)d_ws;
  const size_t szXB = (size_t)SEQLEN * DIM * 2;
  const size_t szWT = (size_t)DIM * DIM * 2;
  __bf16* xb  = (__bf16*)w; w += szXB;
  __bf16* wqT = (__bf16*)w; w += szWT;
  __bf16* wkT = (__bf16*)w; w += szWT;
  __bf16* wvT = (__bf16*)w; w += szWT;
  __bf16* woT = (__bf16*)w; w += szWT;
  __bf16* qb  = (__bf16*)w; w += szXB;
  __bf16* kb  = (__bf16*)w; w += szXB;
  __bf16* vb  = (__bf16*)w; w += szXB;
  __bf16* ab  = (__bf16*)w; w += szXB;
  /* no pad after ab: GEMM clamps A rows or bounds TDM by mvalid, so M-overhang reads stay in bounds */

  const int nx = SEQLEN * DIM;
  cvt_x_kernel<<<(nx + 255) / 256, 256, 0, stream>>>(x, xb, nx);
  cvt_wT_kernel<<<dim3(DIM, 4), 256, 0, stream>>>(Wq, Wk, Wv, Wo, wqT, wkT, wvT, woT);

  dim3 gg((SEQLEN + 63) / 64, DIM / 64);
  gemm_bias_kernel<true><<<gg, 256, 0, stream>>>(xb, wqT, bq, qb);
  gemm_bias_kernel<true><<<gg, 256, 0, stream>>>(xb, wkT, bk, kb);
  gemm_bias_kernel<true><<<gg, 256, 0, stream>>>(xb, wvT, bv, vb);

  norm_rope_kernel<<<dim3(SEQLEN, 2), 256, 0, stream>>>(qb, kb, gq, gk, freqs);

  attn_kernel<<<dim3((SEQLEN + 15) / 16, HEADS), 32, 0, stream>>>(qb, kb, vb, ab);

  gemm_bias_kernel<false><<<gg, 256, 0, stream>>>(ab, woT, bo, out);
}
